// SoftProposal_66331474920015
// MI455X (gfx1250) — matrix operand order repaired and checked by reading, compiled, not run on hardware
//
#include <hip/hip_runtime.h>

#define BATCH 32
#define CDIM  512
#define NPIX  1024
#define NB    16                  // 1024 / 64-wide blocks
#define NTRI  (NB * (NB + 1) / 2) // 136 upper-tri 64x64 blocks
#define NITER 20
#define EPSV  1e-12f

typedef __attribute__((ext_vector_type(16))) __bf16 v16bf;
typedef __attribute__((ext_vector_type(8)))  float  v8f;
typedef unsigned int u32x4 __attribute__((ext_vector_type(4)));
typedef int          i32x4 __attribute__((ext_vector_type(4)));
typedef int          i32x8 __attribute__((ext_vector_type(8)));

struct alignas(16) U128 { unsigned int w[4]; };
union FragBF { U128 q[2]; v16bf v; };

#if defined(__has_builtin)
#if __has_builtin(__builtin_amdgcn_tensor_load_to_lds) && \
    __has_builtin(__builtin_amdgcn_s_wait_tensorcnt)
#define HAVE_TDM 1
#endif
#endif

#ifdef HAVE_TDM
// ---------------------------------------------------------------------------
// TDM: DMA one 64-col x 32-k bf16 slab (u[b] row stride CDIM) into LDS laid
// out as [col][40] ushorts. pad_interval=16 DWORDs (64B row), pad_amount=4
// DWORDs (16B) reproduces the +8-ushort padding natively.
// ---------------------------------------------------------------------------
__device__ __forceinline__ void tdm_load_tile(unsigned lds_addr,
                                              const unsigned short* gptr) {
    unsigned long long ga = (unsigned long long)(uintptr_t)gptr;
    unsigned glo = (unsigned)__builtin_amdgcn_readfirstlane((int)(ga & 0xffffffffu));
    unsigned ghi = (unsigned)__builtin_amdgcn_readfirstlane((int)(ga >> 32));
    unsigned la  = (unsigned)__builtin_amdgcn_readfirstlane((int)lds_addr);

    u32x4 g0;
    g0.x = 1u;                                        // count=1 (valid), user mode
    g0.y = la;                                        // lds_addr (bytes)
    g0.z = glo;                                       // global_addr[31:0]
    g0.w = (ghi & 0x01ffffffu) | (2u << 30);          // global_addr[56:32] | type=2

    i32x8 g1;
    g1[0] = (int)((1u << 16)                          // data_size = 2B
                | (1u << 20)                          // pad_enable
                | (3u << 22)                          // pad_interval: 16 DWORDs
                | (3u << 25));                        // pad_amount:   4 DWORDs
    g1[1] = (int)(512u << 16);                        // tensor_dim0 = 512 (lo16)
    g1[2] = (int)(1024u << 16);                       // dim0 hi | tensor_dim1 = 1024
    g1[3] = (int)(32u << 16);                         // dim1 hi | tile_dim0 = 32 (k)
    g1[4] = 64;                                       // tile_dim1 = 64 (cols), dim2=0
    g1[5] = 512;                                      // tensor_dim0_stride = CDIM
    g1[6] = 0;
    g1[7] = 0;

    i32x4 z4 = {0, 0, 0, 0};
#if __clang_major__ >= 23
    i32x8 z8 = {0, 0, 0, 0, 0, 0, 0, 0};
    __builtin_amdgcn_tensor_load_to_lds(g0, g1, z4, z4, z8, 0);
#else
    __builtin_amdgcn_tensor_load_to_lds(g0, g1, z4, z4, 0);
#endif
}
#endif // HAVE_TDM

// ---------------------------------------------------------------------------
// 1) Spatial Gaussian metric D[i,j] = exp(-d2 / (2*FACTOR*N))
// ---------------------------------------------------------------------------
__global__ void dmat_kernel(float* __restrict__ Dm) {
    int idx = blockIdx.x * 256 + threadIdx.x;       // N*N = 1,048,576
    int i = idx >> 10, j = idx & 1023;
    float dy = (float)((i >> 5) - (j >> 5));
    float dx = (float)((i & 31) - (j & 31));
    Dm[idx] = __expf(-(dy * dy + dx * dx) * (1.0f / 307.2f)); // 2*(0.15*1024)
}

// ---------------------------------------------------------------------------
// 2) sq[b][n] = sum_c x[b][c][n]^2   (f32, coalesced over n)
// ---------------------------------------------------------------------------
__global__ void sq_kernel(const float* __restrict__ x, float* __restrict__ SQ) {
    int idx = blockIdx.x * 256 + threadIdx.x;       // B*N = 32768
    int b = idx >> 10, n = idx & 1023;
    const float* xb = x + (size_t)b * CDIM * NPIX + n;
    float s = 0.f;
    for (int c = 0; c < CDIM; ++c) { float v = xb[(size_t)c * NPIX]; s += v * v; }
    SQ[idx] = s;
}

// ---------------------------------------------------------------------------
// 3) Pack x[b][c][n] (f32) -> U[b][n][c] (bf16, RNE) via LDS 32x32 transpose
// ---------------------------------------------------------------------------
__global__ void pack_kernel(const float* __restrict__ x, unsigned short* __restrict__ U) {
    int blk = blockIdx.x;                 // B * (C/32) * (N/32) = 16384
    int b   = blk / (16 * 32);
    int rem = blk % (16 * 32);
    int cb  = rem >> 5;                   // 0..15
    int nb  = rem & 31;                   // 0..31
    __shared__ unsigned short tile[32][33];
    const float* xb = x + ((size_t)b * CDIM + cb * 32) * NPIX + nb * 32;
    for (int k = 0; k < 4; ++k) {
        int e = threadIdx.x + k * 256;
        int cc = e >> 5, nn = e & 31;
        unsigned u = __float_as_uint(xb[(size_t)cc * NPIX + nn]);
        u += 0x7fffu + ((u >> 16) & 1u);  // round-to-nearest-even to bf16
        tile[cc][nn] = (unsigned short)(u >> 16);
    }
    __syncthreads();
    unsigned short* ub = U + ((size_t)b * NPIX + nb * 32) * CDIM + cb * 32;
    for (int k = 0; k < 4; ++k) {
        int e = threadIdx.x + k * 256;
        int cc = e & 31, nn = e >> 5;
        ub[(size_t)nn * CDIM + cc] = tile[cc][nn];
    }
}

// ---------------------------------------------------------------------------
// 4) Gram via WMMA bf16 + fused fdist/D epilogue -> T (symmetric: write both
//    triangles from one upper-tri 64x64 block). 4 waves, each 32x32 output.
//    Staging: TDM tensor_load_to_lds, double-buffered (fallback: manual).
// ---------------------------------------------------------------------------
__global__ __launch_bounds__(128) void gram_t_kernel(
    const unsigned short* __restrict__ U, const float* __restrict__ SQ,
    const float* __restrict__ Dm, float* __restrict__ T)
{
    int b = blockIdx.x / NTRI;
    int t = blockIdx.x % NTRI;
    int bi = 0;
    while (t >= NB - bi) { t -= NB - bi; ++bi; }
    int bj = bi + t;                      // bi <= bj
    const int i0 = bi * 64, j0 = bj * 64;

    __shared__ __align__(16) unsigned short As[2][64][40];  // [buf][col i][k]
    __shared__ __align__(16) unsigned short Bs[2][64][40];  // [buf][col j][k]
    __shared__ float Ts[64][65];

    const int tid = threadIdx.x;
    const int lane = tid & 31, wave = tid >> 5;
    const int lane16 = lane & 15, hi = lane >> 4;
    const int wi = wave >> 1, wj = wave & 1;             // wave -> 32x32 tile
    const unsigned short* ub = U + (size_t)b * NPIX * CDIM;

    v8f z = {0.f, 0.f, 0.f, 0.f, 0.f, 0.f, 0.f, 0.f};
    v8f acc[2][2];
    acc[0][0] = z; acc[0][1] = z; acc[1][0] = z; acc[1][1] = z;

#ifdef HAVE_TDM
    // ---- TDM path: one DMA per slab, issued by waves 0/1, double-buffered
    const unsigned ldsA0 = (unsigned)(uintptr_t)(void*)&As[0][0][0];
    const unsigned ldsA1 = (unsigned)(uintptr_t)(void*)&As[1][0][0];
    const unsigned ldsB0 = (unsigned)(uintptr_t)(void*)&Bs[0][0][0];
    const unsigned ldsB1 = (unsigned)(uintptr_t)(void*)&Bs[1][0][0];

    if (wave == 0) tdm_load_tile(ldsA0, ub + (size_t)i0 * CDIM);
    if (wave == 1) tdm_load_tile(ldsB0, ub + (size_t)j0 * CDIM);
    __builtin_amdgcn_s_wait_tensorcnt(0);
    __syncthreads();

    for (int kb = 0; kb < CDIM; kb += 32) {
        const int cur = (kb >> 5) & 1;
        if (kb + 32 < CDIM) {             // kick next slab into other buffer
            if (wave == 0)
                tdm_load_tile(cur ? ldsA0 : ldsA1, ub + (size_t)i0 * CDIM + kb + 32);
            if (wave == 1)
                tdm_load_tile(cur ? ldsB0 : ldsB1, ub + (size_t)j0 * CDIM + kb + 32);
        }

        FragBF a[2], bfrag[2];
        for (int mi = 0; mi < 2; ++mi) {
            int col = wi * 32 + mi * 16 + lane16;
            a[mi].q[0] = *(const U128*)&As[cur][col][hi * 8];
            a[mi].q[1] = *(const U128*)&As[cur][col][16 + hi * 8];
        }
        for (int nj = 0; nj < 2; ++nj) {
            int col = wj * 32 + nj * 16 + lane16;
            bfrag[nj].q[0] = *(const U128*)&Bs[cur][col][hi * 8];
            bfrag[nj].q[1] = *(const U128*)&Bs[cur][col][16 + hi * 8];
        }
        for (int mi = 0; mi < 2; ++mi)
            for (int nj = 0; nj < 2; ++nj)
                acc[mi][nj] = __builtin_amdgcn_wmma_f32_16x16x32_bf16(
                    false, a[mi].v, false, bfrag[nj].v,
                    (short)0, acc[mi][nj], false, false);

        __builtin_amdgcn_s_wait_tensorcnt(0);   // issuing waves drain their DMA
        __syncthreads();                        // publish LDS to all waves
    }
#else
    // ---- fallback: cooperative global->VGPR->LDS staging (round-1 path)
    const int tcol = tid >> 1;            // 0..63
    const int toff = (tid & 1) * 16;      // ushort offset 0 / 16 (32B chunks)
    for (int kb = 0; kb < CDIM; kb += 32) {
        const unsigned short* ga = ub + (size_t)(i0 + tcol) * CDIM + kb + toff;
        const unsigned short* gb = ub + (size_t)(j0 + tcol) * CDIM + kb + toff;
        ((U128*)&As[0][tcol][toff])[0] = ((const U128*)ga)[0];
        ((U128*)&As[0][tcol][toff])[1] = ((const U128*)ga)[1];
        ((U128*)&Bs[0][tcol][toff])[0] = ((const U128*)gb)[0];
        ((U128*)&Bs[0][tcol][toff])[1] = ((const U128*)gb)[1];
        __syncthreads();
        if (kb + 32 < CDIM) {
            __builtin_prefetch(ga + 32, 0, 0);
            __builtin_prefetch(gb + 32, 0, 0);
        }
        FragBF a[2], bfrag[2];
        for (int mi = 0; mi < 2; ++mi) {
            int col = wi * 32 + mi * 16 + lane16;
            a[mi].q[0] = *(const U128*)&As[0][col][hi * 8];
            a[mi].q[1] = *(const U128*)&As[0][col][16 + hi * 8];
        }
        for (int nj = 0; nj < 2; ++nj) {
            int col = wj * 32 + nj * 16 + lane16;
            bfrag[nj].q[0] = *(const U128*)&Bs[0][col][hi * 8];
            bfrag[nj].q[1] = *(const U128*)&Bs[0][col][16 + hi * 8];
        }
        for (int mi = 0; mi < 2; ++mi)
            for (int nj = 0; nj < 2; ++nj)
                acc[mi][nj] = __builtin_amdgcn_wmma_f32_16x16x32_bf16(
                    false, a[mi].v, false, bfrag[nj].v,
                    (short)0, acc[mi][nj], false, false);
        __syncthreads();
    }
#endif

    // epilogue: T = sqrt(max(sq_i + sq_j - 2*gram, 0) + eps) * D[i,j]
    const float* sqb = SQ + (b << 10);
    for (int mi = 0; mi < 2; ++mi)
        for (int nj = 0; nj < 2; ++nj) {
            int jl = wj * 32 + nj * 16 + lane16;
            int jg = j0 + jl;
            float sqj = sqb[jg];
            for (int r = 0; r < 8; ++r) {
                int il = wi * 32 + mi * 16 + hi * 8 + r;   // C/D row map
                int ig = i0 + il;
                float g = acc[mi][nj][r];
                float fd2 = fmaxf(sqb[ig] + sqj - 2.0f * g, 0.0f);
                Ts[il][jl] = sqrtf(fd2 + EPSV) * Dm[(size_t)ig * NPIX + jg];
            }
        }
    __syncthreads();

    float* Tb = T + (size_t)b * NPIX * NPIX;
    for (int e = tid; e < 64 * 64; e += 128) {
        int li = e >> 6, lj = e & 63;
        Tb[(size_t)(i0 + li) * NPIX + (j0 + lj)] = Ts[li][lj];
    }
    if (bi != bj) {                       // mirrored block, coalesced rows
        for (int e = tid; e < 64 * 64; e += 128) {
            int li = e >> 6, lj = e & 63;
            Tb[(size_t)(j0 + li) * NPIX + (i0 + lj)] = Ts[lj][li];
        }
    }
}

// ---------------------------------------------------------------------------
// 5) inverse column sums (fold normalization into iteration vector)
// ---------------------------------------------------------------------------
__global__ void colsum_kernel(const float* __restrict__ T, float* __restrict__ CSI) {
    int idx = blockIdx.x * 256 + threadIdx.x;     // B*N
    int b = idx >> 10, j = idx & 1023;
    const float* Tb = T + (size_t)b * NPIX * NPIX + j;
    float s = 0.f;
    for (int i = 0; i < NPIX; ++i) s += Tb[(size_t)i * NPIX];
    CSI[idx] = 1.0f / fmaxf(s, EPSV);
}

__global__ void init_kernel(const float* __restrict__ CSI, float* __restrict__ P,
                            float* __restrict__ W, float* __restrict__ S) {
    int idx = blockIdx.x * 256 + threadIdx.x;     // B*N
    float p0 = 1.0f / (float)NPIX;
    P[idx] = p0;
    W[idx] = p0 * CSI[idx];
    if (idx < NITER * BATCH) S[idx] = 0.f;
}

// ---------------------------------------------------------------------------
// 6) power iteration: r = T*w (bandwidth-bound; T mostly L2-resident)
// ---------------------------------------------------------------------------
__global__ __launch_bounds__(256) void matvec_kernel(
    const float* __restrict__ T, const float* __restrict__ W,
    float* __restrict__ R, float* __restrict__ S, int it)
{
    int lane = threadIdx.x & 31, wave = threadIdx.x >> 5;
    int row = blockIdx.x * 8 + wave;              // B*N rows, one wave/row
    int b = row >> 10;
    const float4* Tr = (const float4*)(T + (size_t)row * NPIX);
    const float4* Wv = (const float4*)(W + (b << 10));
    float s = 0.f;
    for (int q = lane; q < NPIX / 4; q += 32) {
        float4 t = Tr[q], w = Wv[q];
        s += t.x * w.x + t.y * w.y + t.z * w.z + t.w * w.w;
    }
    for (int off = 16; off > 0; off >>= 1) s += __shfl_down(s, off, 32);
    if (lane == 0) { R[row] = s; atomicAdd(&S[it * BATCH + b], s); }
}

__global__ void update_kernel(const float* __restrict__ R, const float* __restrict__ S,
                              const float* __restrict__ CSI, float* __restrict__ P,
                              float* __restrict__ W, int it)
{
    int idx = blockIdx.x * 256 + threadIdx.x;     // B*N
    int b = idx >> 10;
    float p = R[idx] / fmaxf(S[it * BATCH + b], EPSV);
    P[idx] = p;
    W[idx] = p * CSI[idx];
}

// ---------------------------------------------------------------------------
// 7) couple: out = x * proposal (broadcast over channels), float4 streams
// ---------------------------------------------------------------------------
__global__ void couple_kernel(const float* __restrict__ x, const float* __restrict__ P,
                              float* __restrict__ out)
{
    int tid = blockIdx.x * 256 + threadIdx.x;     // (B*C*N)/4 threads
    size_t base = (size_t)tid * 4;
    int b = (int)(base >> 19);                    // / (C*N = 2^19)
    int n = (int)(base & 1023);
    float4 xv = *(const float4*)(x + base);
    float4 pv = *(const float4*)(P + ((size_t)(b << 10) + n));
    float4 ov;
    ov.x = xv.x * pv.x; ov.y = xv.y * pv.y; ov.z = xv.z * pv.z; ov.w = xv.w * pv.w;
    *(float4*)(out + base) = ov;
}

// ---------------------------------------------------------------------------
extern "C" void kernel_launch(void* const* d_in, const int* in_sizes, int n_in,
                              void* d_out, int out_size, void* d_ws, size_t ws_size,
                              hipStream_t stream) {
    (void)in_sizes; (void)n_in; (void)out_size; (void)ws_size;
    const float* x = (const float*)d_in[0];
    float* out = (float*)d_out;

    char* ws = (char*)d_ws;
    size_t off = 0;
    auto alloc = [&](size_t bytes) {
        char* p = ws + off;
        off = (off + bytes + 255) & ~(size_t)255;
        return p;
    };
    float*          Dm  = (float*)alloc((size_t)NPIX * NPIX * 4);            //   4 MB
    unsigned short* U   = (unsigned short*)alloc((size_t)BATCH * NPIX * CDIM * 2); // 32 MB
    float*          SQ  = (float*)alloc((size_t)BATCH * NPIX * 4);
    float*          CSI = (float*)alloc((size_t)BATCH * NPIX * 4);
    float*          P   = (float*)alloc((size_t)BATCH * NPIX * 4);
    float*          R   = (float*)alloc((size_t)BATCH * NPIX * 4);
    float*          W   = (float*)alloc((size_t)BATCH * NPIX * 4);
    float*          S   = (float*)alloc((size_t)NITER * BATCH * 4);
    float*          T   = (float*)alloc((size_t)BATCH * NPIX * NPIX * 4);    // 128 MB

    dmat_kernel<<<(NPIX * NPIX) / 256, 256, 0, stream>>>(Dm);
    sq_kernel<<<(BATCH * NPIX) / 256, 256, 0, stream>>>(x, SQ);
    pack_kernel<<<BATCH * 16 * 32, 256, 0, stream>>>(x, U);
    gram_t_kernel<<<BATCH * NTRI, 128, 0, stream>>>(U, SQ, Dm, T);
    colsum_kernel<<<(BATCH * NPIX) / 256, 256, 0, stream>>>(T, CSI);
    init_kernel<<<(BATCH * NPIX) / 256, 256, 0, stream>>>(CSI, P, W, S);
    for (int it = 0; it < NITER; ++it) {
        matvec_kernel<<<(BATCH * NPIX) / 8, 256, 0, stream>>>(T, W, R, S, it);
        update_kernel<<<(BATCH * NPIX) / 256, 256, 0, stream>>>(R, S, CSI, P, W, it);
    }
    couple_kernel<<<(BATCH * CDIM * NPIX / 4) / 256, 256, 0, stream>>>(x, P, out);
}
